// GPT_10256381902898
// MI455X (gfx1250) — matrix operand order repaired and checked by reading, not compiled, not run on hardware
//
#include <hip/hip_runtime.h>
#include <math.h>

// ---------------------------------------------------------------------------
// GPT-2 (124M) forward for gfx1250 (MI455X).
//  - All GEMMs AND attention: v_wmma_f32_16x16x32_bf16, fp32 accumulation.
//  - Weights pre-converted fp32->bf16 and transposed to [N,K] once per call.
//  - GEMM tiles staged global->LDS with TENSOR_LOAD_TO_LDS (TDM), double
//    buffered, padded in-flight (stride 36 bf16) to avoid LDS bank conflicts.
//  - Attention: WMMA flash-attention, 64-query blocks, online softmax.
// ---------------------------------------------------------------------------

#define NB   2
#define NT   1024
#define ND   768
#define NH   12
#define NHD  64
#define NL   12
#define NFF  3072
#define NV   50257
#define NM   (NB * NT)          // 2048 token rows

typedef __attribute__((ext_vector_type(16))) __bf16 bf16x16;
typedef __attribute__((ext_vector_type(8)))  float  floatx8;

enum { EPI_BIAS = 0, EPI_RES = 1, EPI_GELU = 2, EPI_NONE = 3 };

#if defined(__HIP_DEVICE_COMPILE__) && \
    __has_builtin(__builtin_amdgcn_tensor_load_to_lds) && \
    __has_builtin(__builtin_amdgcn_s_wait_tensorcnt)
#define HAVE_TDM 1
#else
#define HAVE_TDM 0
#endif

// ----------------------------- helpers -------------------------------------

__device__ __forceinline__ unsigned short f2bf(float f) {
  union { float f; unsigned u; } c; c.f = f;
  unsigned u = c.u;
  u += 0x7fffu + ((u >> 16) & 1u);   // round-to-nearest-even
  return (unsigned short)(u >> 16);
}

__device__ __forceinline__ __bf16 us2bf(unsigned short s) {
  union { unsigned short s; __bf16 b; } c; c.s = s;
  return c.b;
}

__device__ __forceinline__ float mgelu(float v) {
  const float c = 0.7978845608028654f;              // sqrt(2/pi)
  float t = tanhf(c * (v + 0.047715f * v * v * v)); // module's 0.047715
  return 0.5f * v * (1.0f + t);
}

__device__ __forceinline__ float wave_sum(float v) {
#pragma unroll
  for (int off = 16; off >= 1; off >>= 1) v += __shfl_xor(v, off, 32);
  return v;
}

__device__ __forceinline__ float wave_max(float v) {
#pragma unroll
  for (int off = 16; off >= 1; off >>= 1) v = fmaxf(v, __shfl_xor(v, off, 32));
  return v;
}

// 16-lane (half-wave) reductions: xor masks 1..8 stay within a 16-lane half.
__device__ __forceinline__ float half_max(float v) {
#pragma unroll
  for (int off = 8; off >= 1; off >>= 1) v = fmaxf(v, __shfl_xor(v, off, 32));
  return v;
}
__device__ __forceinline__ float half_sum(float v) {
#pragma unroll
  for (int off = 8; off >= 1; off >>= 1) v += __shfl_xor(v, off, 32);
  return v;
}

// ------------------------- WMMA fragment loaders ----------------------------
// A fragment (16x32, MxK): lane l15 = row; halves split K 0..15 / 16..31.
__device__ __forceinline__ bf16x16 load_afrag(
    const unsigned short* base, int stride, int row, int koff, int half) {
  bf16x16 a;
#pragma unroll
  for (int e = 0; e < 8; ++e) {
    a[e]     = us2bf(base[row * stride + koff + half * 16 + e]);
    a[e + 8] = us2bf(base[row * stride + koff + half * 16 + 8 + e]);
  }
  return a;
}
// B fragment (32x16, KxN): lane l15 = column; halves split K 0..15 / 16..31.
__device__ __forceinline__ bf16x16 load_bfrag(
    const unsigned short* base, int stride, int col, int koff, int half) {
  bf16x16 b;
#pragma unroll
  for (int e = 0; e < 16; ++e)
    b[e] = us2bf(base[col * stride + koff + half * 16 + e]);
  return b;
}
__device__ __forceinline__ floatx8 wmma_bf16(bf16x16 a, bf16x16 b, floatx8 c) {
  return __builtin_amdgcn_wmma_f32_16x16x32_bf16(
      false, a, false, b, (short)0, c, false, false);
}

// ----------------------------- TDM tile load --------------------------------
// 2D tile load: tile_dim0 = 32 bf16 (64B rows), tile_dim1 = 128 rows,
// row stride = stride_elems (bf16 units) in global. LDS destination gets
// 2 dwords of padding after every 16 dwords -> effective LDS row stride of
// 36 bf16, matching the fragment readers.

#if HAVE_TDM
typedef __attribute__((ext_vector_type(4))) unsigned int uintx4;
typedef __attribute__((ext_vector_type(8))) int intx8;
typedef __attribute__((ext_vector_type(4))) int intx4;

__device__ __forceinline__ void tdm_load_tile(
    unsigned lds_byte, const unsigned short* gptr,
    int rows_remaining, int k_remaining, int stride_elems) {
  unsigned long long ga = (unsigned long long)gptr;
  uintx4 g0;
  g0.x = 1u;                                         // count=1 (valid, user mode)
  g0.y = lds_byte;                                   // lds_addr
  g0.z = (unsigned)ga;                               // global_addr[31:0]
  g0.w = (unsigned)((ga >> 32) & 0x01FFFFFFull)      // global_addr[56:32]
         | (2u << 30);                               // type = 2 ("image")
  unsigned td0 = (unsigned)k_remaining;              // tensor_dim0 (elems)
  unsigned td1 = (unsigned)rows_remaining;           // tensor_dim1 (rows)
  intx8 g1;
  g1[0] = (int)((1u << 16)                           // data_size = 2 bytes
                | (1u << 20)                         // pad_enable
                | (3u << 22)                         // pad_interval: 16 dwords
                | (1u << 25));                       // pad_amount: 2 dwords
  g1[1] = (int)((td0 & 0xFFFFu) << 16);              // tensor_dim0[15:0]
  g1[2] = (int)(((td0 >> 16) & 0xFFFFu) | ((td1 & 0xFFFFu) << 16));
  g1[3] = (int)(((td1 >> 16) & 0xFFFFu) | (32u << 16));  // tile_dim0 = 32
  g1[4] = 128;                                       // tile_dim1 = 128, tile_dim2 = 0
  g1[5] = (int)(unsigned)stride_elems;               // tensor_dim0_stride[31:0]
  g1[6] = 0;                                         // stride[47:32], dim1_stride lo
  g1[7] = 0;
  intx4 z4 = {0, 0, 0, 0};
#if __clang_major__ >= 23
  intx8 z8 = {0, 0, 0, 0, 0, 0, 0, 0};
  __builtin_amdgcn_tensor_load_to_lds(g0, g1, z4, z4, z8, 0);
#else
  __builtin_amdgcn_tensor_load_to_lds(g0, g1, z4, z4, 0);
#endif
}
#endif  // HAVE_TDM

// ----------------------------- embedding -----------------------------------

__global__ __launch_bounds__(256) void embed_kernel(
    const int* __restrict__ idx, const float* __restrict__ wte,
    const float* __restrict__ wpe, float* __restrict__ x) {
  int e = blockIdx.x * 256 + threadIdx.x;   // < NM*ND
  int row = e / ND;
  int c   = e - row * ND;
  int t   = row & (NT - 1);
  x[e] = wte[(size_t)idx[row] * ND + c] + wpe[(size_t)t * ND + c];
}

// ----------------------------- layernorm (fp32 in, bf16 out) ---------------

__global__ __launch_bounds__(256) void ln_kernel(
    const float* __restrict__ x, const float* __restrict__ g,
    const float* __restrict__ b, unsigned short* __restrict__ out) {
  int row  = blockIdx.x;
  int tid  = threadIdx.x;
  int lane = tid & 31, wave = tid >> 5;
  const float* xr = x + (size_t)row * ND;
  unsigned short* outr = out + (size_t)row * ND;

  float v0 = xr[tid], v1 = xr[tid + 256], v2 = xr[tid + 512];
  float s  = v0 + v1 + v2;
  float s2 = v0 * v0 + v1 * v1 + v2 * v2;
  s = wave_sum(s); s2 = wave_sum(s2);

  __shared__ float shs[8], shs2[8];
  if (lane == 0) { shs[wave] = s; shs2[wave] = s2; }
  __syncthreads();
  float S = 0.f, S2 = 0.f;
#pragma unroll
  for (int i = 0; i < 8; ++i) { S += shs[i]; S2 += shs2[i]; }

  float mean = S * (1.0f / ND);
  float var  = S2 * (1.0f / ND) - mean * mean;
  float rstd = rsqrtf(var + 1e-5f);

  outr[tid]       = f2bf((v0 - mean) * rstd * g[tid]       + b[tid]);
  outr[tid + 256] = f2bf((v1 - mean) * rstd * g[tid + 256] + b[tid + 256]);
  outr[tid + 512] = f2bf((v2 - mean) * rstd * g[tid + 512] + b[tid + 512]);
}

// ------------------- weight convert + transpose (fp32[K,N] -> bf16[N,K]) ---

__global__ __launch_bounds__(256) void wconv_kernel(
    const float* __restrict__ src, unsigned short* __restrict__ dst,
    int K, int N) {
  __shared__ unsigned short t[32][33];
  int k0 = blockIdx.x * 32, n0 = blockIdx.y * 32;
  int tx = threadIdx.x & 31, ty = threadIdx.x >> 5;   // ty in 0..7
#pragma unroll
  for (int i = 0; i < 4; ++i) {
    int kk = ty + i * 8;
    int n  = n0 + tx;
    float f = (n < N) ? src[(size_t)(k0 + kk) * N + n] : 0.0f;
    t[kk][tx] = f2bf(f);
  }
  __syncthreads();
#pragma unroll
  for (int i = 0; i < 4; ++i) {
    int nn = ty + i * 8;
    int n  = n0 + nn;
    if (n < N) dst[(size_t)n * K + k0 + tx] = t[tx][nn];
  }
}

// ----------------------------- WMMA GEMM -----------------------------------
// C[M,N] = A[M,K] @ Wt[N,K]^T (+bias) (+residual | gelu).
// A, Wt: bf16 in global. LDS tiles: [128][36] bf16 each, double buffered,
// filled by TDM. Block: 256 threads = 8 waves (2x4); 4x2 WMMA tiles per wave.

#define BM 128
#define BN 128
#define BK 32
#define LDSP 36                       // padded LDS row stride (bf16 units)
#define TILE_BYTES (128 * LDSP * 2)   // 9216 B per tile
#define BUF_BYTES  (2 * TILE_BYTES)   // A + B per buffer
#define SMEM_BYTES (2 * BUF_BYTES)    // double buffered: 36864 B

__device__ __forceinline__ void wmma_step(
    const unsigned short* sA, const unsigned short* sB,
    floatx8 acc[4][2], int wm, int wn, int l15, int half) {
  bf16x16 bfrag[2];
#pragma unroll
  for (int j = 0; j < 2; ++j)
    bfrag[j] = load_bfrag(sB, LDSP, wn * 32 + j * 16 + l15, 0, half);
#pragma unroll
  for (int i = 0; i < 4; ++i) {
    bf16x16 afrag = load_afrag(sA, LDSP, wm * 64 + i * 16 + l15, 0, half);
#pragma unroll
    for (int j = 0; j < 2; ++j)
      acc[i][j] = wmma_bf16(afrag, bfrag[j], acc[i][j]);
  }
}

template <int EPI, int OUTBF>
__global__ __launch_bounds__(256) void gemm_kernel(
    const unsigned short* __restrict__ A, const unsigned short* __restrict__ Wt,
    const float* __restrict__ bias, const float* residual,
    void* Cv, int M, int N, int K) {
  extern __shared__ char smem[];   // SMEM_BYTES, dynamic => LDS offset 0

  int tid  = threadIdx.x;
  int wave = tid >> 5;
  int lane = tid & 31;
  int wm   = wave >> 2;       // 0..1
  int wn   = wave & 3;        // 0..3
  int l15  = lane & 15;
  int half = lane >> 4;
  int m0   = blockIdx.y * BM;
  int n0   = blockIdx.x * BN;

  floatx8 acc[4][2];
  const floatx8 zf = {0.f, 0.f, 0.f, 0.f, 0.f, 0.f, 0.f, 0.f};
#pragma unroll
  for (int i = 0; i < 4; ++i)
#pragma unroll
    for (int j = 0; j < 2; ++j) acc[i][j] = zf;

  const int ktiles = K / BK;

#if HAVE_TDM
  // ---- TDM double-buffered pipeline ----
  if (wave == 0) {
    tdm_load_tile(0,          A  + (size_t)m0 * K, M - m0, K, K);
    tdm_load_tile(TILE_BYTES, Wt + (size_t)n0 * K, N - n0, K, K);
    __builtin_amdgcn_s_wait_tensorcnt((short)0);
  }
  __syncthreads();

  for (int kt = 0; kt < ktiles; ++kt) {
    int cur = kt & 1;
    if (wave == 0 && kt + 1 < ktiles) {
      int k1 = (kt + 1) * BK;
      int nxt = cur ^ 1;
      tdm_load_tile((unsigned)(nxt * BUF_BYTES),
                    A + (size_t)m0 * K + k1, M - m0, K - k1, K);
      tdm_load_tile((unsigned)(nxt * BUF_BYTES + TILE_BYTES),
                    Wt + (size_t)n0 * K + k1, N - n0, K - k1, K);
    }
    const unsigned short* sA =
        (const unsigned short*)(smem + cur * BUF_BYTES);
    const unsigned short* sB =
        (const unsigned short*)(smem + cur * BUF_BYTES + TILE_BYTES);
    wmma_step(sA, sB, acc, wm, wn, l15, half);
    if (wave == 0) __builtin_amdgcn_s_wait_tensorcnt((short)0);
    __syncthreads();
  }
#else
  // ---- fallback: cooperative staging, single buffer ----
  unsigned short* sA = (unsigned short*)smem;
  unsigned short* sB = (unsigned short*)(smem + TILE_BYTES);
  for (int kt = 0; kt < ktiles; ++kt) {
    int k0 = kt * BK;
#pragma unroll
    for (int i = 0; i < 16; ++i) {
      int e = i * 256 + tid;
      int r = e >> 5, c = e & 31;
      sA[r * LDSP + c] = A[(size_t)(m0 + r) * K + k0 + c];
      int gn = n0 + r;
      sB[r * LDSP + c] = (gn < N) ? Wt[(size_t)gn * K + k0 + c]
                                  : (unsigned short)0;
    }
    __syncthreads();
    wmma_step(sA, sB, acc, wm, wn, l15, half);
    __syncthreads();
  }
#endif

  // Epilogue. C/D layout: VGPR r -> M = r + 8*half; lane l15 -> N column.
#pragma unroll
  for (int i = 0; i < 4; ++i) {
    int mbase = m0 + wm * 64 + i * 16 + half * 8;
#pragma unroll
    for (int j = 0; j < 2; ++j) {
      int n = n0 + wn * 32 + j * 16 + l15;
      if (n < N) {
        float bv = (EPI == EPI_NONE) ? 0.0f : bias[n];
#pragma unroll
        for (int r = 0; r < 8; ++r) {
          int m = mbase + r;
          size_t off = (size_t)m * N + n;
          float cv = acc[i][j][r] + bv;
          if (EPI == EPI_RES)  cv += residual[off];
          if (EPI == EPI_GELU) cv = mgelu(cv);
          if (OUTBF) ((unsigned short*)Cv)[off] = f2bf(cv);
          else       ((float*)Cv)[off] = cv;
        }
      }
    }
  }
}

// ------------------------ WMMA flash attention ------------------------------
// One workgroup (8 waves) per (b, head, 64-query block). K/V processed in
// 64-key blocks with online softmax. S = Q.K^T and O += P.V on WMMA.
// Wave (qt, grp): qt = wave>>1 owns q rows qt*16..+15; grp = wave&1 owns
// S/P key-tiles {2grp,2grp+1} and O hd-tiles {2grp,2grp+1}.

#define APAD 68   // LDS row stride for 64-wide attention tiles

__global__ __launch_bounds__(256) void attn_kernel(
    const float* __restrict__ qkv, unsigned short* __restrict__ y) {
  __shared__ unsigned short sQ[64 * APAD];
  __shared__ unsigned short sK[64 * APAD];
  __shared__ unsigned short sV[64 * APAD];   // stored transposed: [hd][key]
  __shared__ unsigned short sP[64 * APAD];
  __shared__ float smax[4][2][16];
  __shared__ float ssum[4][2][16];

  int tid  = threadIdx.x;
  int wave = tid >> 5;
  int lane = tid & 31;
  int qt   = wave >> 1;       // 0..3
  int grp  = wave & 1;        // 0..1
  int l15  = lane & 15;
  int half = lane >> 4;

  int qb = blockIdx.x & 15;                 // 16 query blocks per (b,h)
  int hh = (blockIdx.x >> 4) % NH;
  int b  = blockIdx.x / (16 * NH);
  int qbase = qb * 64;
  int bT    = b * NT;
  int hh0   = hh * NHD;

  // Stage Q block (fp32 -> bf16), coalesced.
#pragma unroll
  for (int i = 0; i < 16; ++i) {
    int e = i * 256 + tid;
    int r = e >> 6, c = e & 63;
    sQ[r * APAD + c] =
        f2bf(qkv[(size_t)(bT + qbase + r) * (3 * ND) + hh0 + c]);
  }
  __syncthreads();

  bf16x16 qfrag[2];
#pragma unroll
  for (int h2 = 0; h2 < 2; ++h2)
    qfrag[h2] = load_afrag(sQ, APAD, qt * 16 + l15, h2 * 32, half);

  floatx8 O[2];
  const floatx8 zf = {0.f, 0.f, 0.f, 0.f, 0.f, 0.f, 0.f, 0.f};
  O[0] = zf; O[1] = zf;
  float m_run[8], denom[8];
#pragma unroll
  for (int r = 0; r < 8; ++r) { m_run[r] = -3.0e38f; denom[r] = 0.f; }

  for (int kb = 0; kb <= qb; ++kb) {
    int kbase = kb * 64;
    __syncthreads();   // protect sK/sV/sP from previous iteration readers
    // Stage K [key][hd] and V transposed [hd][key].
#pragma unroll
    for (int i = 0; i < 16; ++i) {
      int e = i * 256 + tid;
      int r = e >> 6, c = e & 63;
      size_t rowk = (size_t)(bT + kbase + r) * (3 * ND) + ND + hh0;
      sK[r * APAD + c] = f2bf(qkv[rowk + c]);
      sV[c * APAD + r] = f2bf(qkv[rowk + ND + c]);
    }
    __syncthreads();

    // S = Q.K^T for this wave's two key tiles (fp32, scaled, masked).
    floatx8 S[2];
#pragma unroll
    for (int jt = 0; jt < 2; ++jt) {
      floatx8 s = zf;
#pragma unroll
      for (int h2 = 0; h2 < 2; ++h2) {
        bf16x16 kf = load_bfrag(sK, APAD, (grp * 2 + jt) * 16 + l15,
                                h2 * 32, half);
        s = wmma_bf16(qfrag[h2], kf, s);
      }
      int kg = kbase + (grp * 2 + jt) * 16 + l15;
#pragma unroll
      for (int r = 0; r < 8; ++r) {
        float sv = s[r] * 0.125f;                    // 1/sqrt(64)
        int qg = qbase + qt * 16 + r + 8 * half;
        if (kb == qb && kg > qg) sv = -3.0e38f;      // causal mask
        s[r] = sv;
      }
      S[jt] = s;
    }

    // Row max (in-wave 16-lane reduce, then cross-wave via LDS).
    float rmax[8];
#pragma unroll
    for (int r = 0; r < 8; ++r)
      rmax[r] = half_max(fmaxf(S[0][r], S[1][r]));
    if (l15 == 0) {
#pragma unroll
      for (int r = 0; r < 8; ++r) smax[qt][grp][r + 8 * half] = rmax[r];
    }
    __syncthreads();

    // P = exp(S - m_new); store to sP (bf16); row sums.
    float mnew[8], psum[8];
#pragma unroll
    for (int r = 0; r < 8; ++r) {
      int m = r + 8 * half;
      float bm = fmaxf(smax[qt][0][m], smax[qt][1][m]);
      mnew[r] = fmaxf(m_run[r], bm);
    }
#pragma unroll
    for (int jt = 0; jt < 2; ++jt) {
#pragma unroll
      for (int r = 0; r < 8; ++r) {
        float p = __expf(S[jt][r] - mnew[r]);
        S[jt][r] = p;
        sP[(qt * 16 + r + 8 * half) * APAD + (grp * 2 + jt) * 16 + l15] =
            f2bf(p);
      }
    }
#pragma unroll
    for (int r = 0; r < 8; ++r)
      psum[r] = half_sum(S[0][r] + S[1][r]);
    if (l15 == 0) {
#pragma unroll
      for (int r = 0; r < 8; ++r) ssum[qt][grp][r + 8 * half] = psum[r];
    }
    __syncthreads();

    // Denominator update + rescale O, then O += P.V.
    float crr[8];
#pragma unroll
    for (int r = 0; r < 8; ++r) {
      int m = r + 8 * half;
      float bs = ssum[qt][0][m] + ssum[qt][1][m];
      float corr = __expf(m_run[r] - mnew[r]);
      denom[r] = denom[r] * corr + bs;
      m_run[r] = mnew[r];
      crr[r] = corr;
    }
#pragma unroll
    for (int ot = 0; ot < 2; ++ot)
#pragma unroll
      for (int r = 0; r < 8; ++r) O[ot][r] *= crr[r];
#pragma unroll
    for (int kk = 0; kk < 2; ++kk) {
      bf16x16 pf = load_afrag(sP, APAD, qt * 16 + l15, kk * 32, half);
#pragma unroll
      for (int ot = 0; ot < 2; ++ot) {
        bf16x16 vf = load_bfrag(sV, APAD, (grp * 2 + ot) * 16 + l15,
                                kk * 32, half);
        O[ot] = wmma_bf16(pf, vf, O[ot]);
      }
    }
  }

  // Write y (bf16): rows m = r + 8*half, col = hd tile * 16 + l15.
#pragma unroll
  for (int ot = 0; ot < 2; ++ot) {
    int ht = grp * 2 + ot;
#pragma unroll
    for (int r = 0; r < 8; ++r) {
      int qg = qbase + qt * 16 + r + 8 * half;
      y[(size_t)(bT + qg) * ND + hh0 + ht * 16 + l15] =
          f2bf(O[ot][r] / denom[r]);
    }
  }
}

// ----------------------------- loss ----------------------------------------

__global__ __launch_bounds__(256) void loss_row_kernel(
    const float* __restrict__ logits, const int* __restrict__ targets,
    float* __restrict__ nll, float* __restrict__ cnt) {
  int row  = blockIdx.x;
  int tid  = threadIdx.x;
  int lane = tid & 31, wave = tid >> 5;
  const float* lr = logits + (size_t)row * NV;
  __shared__ float sh[8];

  float mx = -3.0e38f;
  for (int c = tid; c < NV; c += 256) mx = fmaxf(mx, lr[c]);
  mx = wave_max(mx);
  if (lane == 0) sh[wave] = mx;
  __syncthreads();
  float m = sh[0];
#pragma unroll
  for (int i = 1; i < 8; ++i) m = fmaxf(m, sh[i]);
  __syncthreads();

  float se = 0.f;
  for (int c = tid; c < NV; c += 256) se += __expf(lr[c] - m);
  se = wave_sum(se);
  if (lane == 0) sh[wave] = se;
  __syncthreads();
  if (tid == 0) {
    float S = 0.f;
#pragma unroll
    for (int i = 0; i < 8; ++i) S += sh[i];
    int tg = targets[row];
    bool valid = tg >= 0;
    nll[row] = valid ? (m + logf(S) - lr[tg]) : 0.f;
    cnt[row] = valid ? 1.f : 0.f;
  }
}

__global__ __launch_bounds__(256) void loss_reduce_kernel(
    const float* __restrict__ nll, const float* __restrict__ cnt,
    float* __restrict__ loss_out) {
  int tid = threadIdx.x, lane = tid & 31, wave = tid >> 5;
  float s = 0.f, c = 0.f;
  for (int i = tid; i < NM; i += 256) { s += nll[i]; c += cnt[i]; }
  s = wave_sum(s); c = wave_sum(c);
  __shared__ float shs[8], shc[8];
  if (lane == 0) { shs[wave] = s; shc[wave] = c; }
  __syncthreads();
  if (tid == 0) {
    float S = 0.f, C = 0.f;
#pragma unroll
    for (int i = 0; i < 8; ++i) { S += shs[i]; C += shc[i]; }
    *loss_out = S / C;
  }
}

// ----------------------------- launch --------------------------------------

extern "C" void kernel_launch(void* const* d_in, const int* in_sizes, int n_in,
                              void* d_out, int out_size, void* d_ws, size_t ws_size,
                              hipStream_t stream) {
  (void)in_sizes; (void)n_in; (void)out_size; (void)ws_size;

  const int*   idx     = (const int*)  d_in[0];
  const int*   targets = (const int*)  d_in[1];
  const float* wte     = (const float*)d_in[2];
  const float* wpe     = (const float*)d_in[3];
  const float* ln1_g   = (const float*)d_in[4];
  const float* ln1_b   = (const float*)d_in[5];
  const float* qkv_w   = (const float*)d_in[6];
  const float* qkv_b   = (const float*)d_in[7];
  const float* po_w    = (const float*)d_in[8];
  const float* po_b    = (const float*)d_in[9];
  const float* ln2_g   = (const float*)d_in[10];
  const float* ln2_b   = (const float*)d_in[11];
  const float* fc_w    = (const float*)d_in[12];
  const float* fc_b    = (const float*)d_in[13];
  const float* pr_w    = (const float*)d_in[14];
  const float* pr_b    = (const float*)d_in[15];
  const float* lnf_g   = (const float*)d_in[16];
  const float* lnf_b   = (const float*)d_in[17];
  const float* lm_w    = (const float*)d_in[18];

  // Workspace layout (byte offsets, 256B-aligned)
  char* ws = (char*)d_ws;
  float*          x    = (float*)(ws);                        //  6,291,456 B
  float*          qkv  = (float*)(ws + 6291456);              // 18,874,368 B
  float*          nll  = (float*)(ws + 25165824);             //      8,192 B
  float*          cnt  = (float*)(ws + 25174016);             //      8,192 B
  unsigned short* h    = (unsigned short*)(ws + 25182208);    //  3,145,728 B
  unsigned short* y    = (unsigned short*)(ws + 28327936);    //  3,145,728 B
  unsigned short* g    = (unsigned short*)(ws + 31473664);    // 12,582,912 B
  unsigned short* wbuf = (unsigned short*)(ws + 44056576);    // 77,194,752 B

  float* logits = (float*)d_out;                  // [NM, NV] then loss scalar

  embed_kernel<<<(NM * ND) / 256, 256, 0, stream>>>(idx, wte, wpe, x);

  const dim3 blk(256);
  for (int l = 0; l < NL; ++l) {
    // ---- attention sub-block ----
    ln_kernel<<<NM, blk, 0, stream>>>(x, ln1_g + l * ND, ln1_b + l * ND, h);
    wconv_kernel<<<dim3(ND / 32, (3 * ND) / 32), blk, 0, stream>>>(
        qkv_w + (size_t)l * ND * 3 * ND, wbuf, ND, 3 * ND);
    gemm_kernel<EPI_BIAS, 0>
        <<<dim3(3 * ND / BN, NM / BM), blk, SMEM_BYTES, stream>>>(
        h, wbuf, qkv_b + (size_t)l * 3 * ND, nullptr, qkv, NM, 3 * ND, ND);
    attn_kernel<<<NB * NH * (NT / 64), blk, 0, stream>>>(qkv, y);
    wconv_kernel<<<dim3(ND / 32, ND / 32), blk, 0, stream>>>(
        po_w + (size_t)l * ND * ND, wbuf, ND, ND);
    gemm_kernel<EPI_RES, 0>
        <<<dim3(ND / BN, NM / BM), blk, SMEM_BYTES, stream>>>(
        y, wbuf, po_b + (size_t)l * ND, x, x, NM, ND, ND);
    // ---- MLP sub-block ----
    ln_kernel<<<NM, blk, 0, stream>>>(x, ln2_g + l * ND, ln2_b + l * ND, h);
    wconv_kernel<<<dim3(ND / 32, NFF / 32), blk, 0, stream>>>(
        fc_w + (size_t)l * ND * NFF, wbuf, ND, NFF);
    gemm_kernel<EPI_GELU, 1>
        <<<dim3(NFF / BN, NM / BM), blk, SMEM_BYTES, stream>>>(
        h, wbuf, fc_b + (size_t)l * NFF, nullptr, g, NM, NFF, ND);
    wconv_kernel<<<dim3(NFF / 32, ND / 32), blk, 0, stream>>>(
        pr_w + (size_t)l * NFF * ND, wbuf, NFF, ND);
    gemm_kernel<EPI_RES, 0>
        <<<dim3(ND / BN, NM / BM), blk, SMEM_BYTES, stream>>>(
        g, wbuf, pr_b + (size_t)l * ND, x, x, NM, ND, NFF);
  }

  ln_kernel<<<NM, blk, 0, stream>>>(x, lnf_g, lnf_b, h);
  wconv_kernel<<<dim3(ND / 32, (NV + 31) / 32), blk, 0, stream>>>(
      lm_w, wbuf, ND, NV);
  gemm_kernel<EPI_NONE, 0>
      <<<dim3((NV + BN - 1) / BN, NM / BM), blk, SMEM_BYTES, stream>>>(
      h, wbuf, nullptr, nullptr, logits, NM, NV, ND);

  loss_row_kernel<<<NM, blk, 0, stream>>>(logits, targets, nll, cnt);
  loss_reduce_kernel<<<1, blk, 0, stream>>>(nll, cnt, logits + (size_t)NM * NV);
}
